// LSTM_GCN_3496103379265
// MI455X (gfx1250) — matrix operand order repaired and checked by reading, not compiled, not run on hardware
//
#include <hip/hip_runtime.h>
#include <hip/hip_bf16.h>
#include <math.h>

// ---------------- problem constants ----------------
#define NB   32              // batch N
#define CINC 64              // input channels
#define TT   64              // timesteps
#define VV   25              // graph nodes
#define VP   32              // padded nodes (full WMMA tiles, zero-padded)
#define HSZ  128             // hidden size
#define SS   8               // graph scales
#define CC1  32
#define CC2  64
#define KX   (SS*CINC)       // 512  (K of input GCN GEMM)
#define KH   (SS*HSZ)        // 1024 (K of recurrent GCN GEMM)
#define MG   (4*HSZ)         // 512  (gate rows)
#define COLS (NB*VP)         // 1024 (padded n*v columns)

typedef __attribute__((ext_vector_type(16))) __bf16 v16bf;
typedef __attribute__((ext_vector_type(8)))  float  v8f;

// ---------------------------------------------------------------------------
// Fragment-native tile storage ("swizzled"):
// every 16x32 A-tile / 32x16 B-tile is 512 contiguous bf16, ordered so that
// lane L owns bytes [L*32, L*32+32) == its v16bf fragment. A wave loads one
// fragment as two coalesced global_load_b128 (or ds_load_b128 from LDS).
//
// Within-lane element mapping (CDNA5 ISA VGPR layouts, wave32):
//  A (16x32, MxK): mlocal = lane&15, half = lane>>4,
//     e -> vg=e>>1, p=e&1, kk = vg<4 ? vg*2+p+half*8 : 16+(vg-4)*2+p+half*8
//  B (32x16, KxN): nlocal = lane&15, half = lane>>4,
//     e -> kk = e<8 ? e+half*8 : 16+(e-8)+half*8
// ---------------------------------------------------------------------------

__device__ __forceinline__ int a_kk_of(int e, int half) {
  const int vg = e >> 1, p = e & 1;
  return (vg < 4) ? (vg * 2 + p + half * 8) : (16 + (vg - 4) * 2 + p + half * 8);
}
__device__ __forceinline__ int b_kk_of(int e, int half) {
  return (e < 8) ? (e + half * 8) : (16 + (e - 8) + half * 8);
}

// load one fragment from a tile-swizzled matrix (global or generic)
__device__ __forceinline__ v16bf load_frag(const __bf16* __restrict__ M, int tile) {
  return *(const v16bf*)(M + ((size_t)tile << 9) + ((threadIdx.x & 31) << 4));
}
__device__ __forceinline__ v16bf load_frag_lds(const __bf16* M, int tile) {
  return *(const v16bf*)(M + (tile << 9) + ((threadIdx.x & 31) << 4));
}

__device__ __forceinline__ float sigmoidf_(float x) { return 1.f / (1.f + expf(-x)); }

// ---------------- K0: weight conversion + bias folding (swizzled output) ----
__global__ void prep_kernel(const float* __restrict__ Wg_w, const float* __restrict__ Ug_w,
                            const float* __restrict__ Wg_b, const float* __restrict__ Ug_b,
                            const float* __restrict__ bb,
                            __bf16* __restrict__ Wg_bf, __bf16* __restrict__ Ug_bf,
                            float* __restrict__ bias_total) {
  const int tid = blockIdx.x * 256 + threadIdx.x;   // grid covers MG*KH = 524288
  {
    // Ug_bf swizzled: tiles [mt(32)][kt(32)] of Ug_w (512 x 1024)
    const int tile = tid >> 9, within = tid & 511;
    const int mt = tile >> 5, kt = tile & 31;
    const int lane = within >> 4, e = within & 15;
    const int half = lane >> 4;
    const int m = mt * 16 + (lane & 15);
    const int k = kt * 32 + a_kk_of(e, half);
    Ug_bf[tid] = (__bf16)Ug_w[(size_t)m * KH + k];
  }
  if (tid < MG * KX) {
    // Wg_bf swizzled: tiles [mt(32)][kt(16)] of Wg_w (512 x 512)
    const int tile = tid >> 9, within = tid & 511;
    const int mt = tile >> 4, kt = tile & 15;
    const int lane = within >> 4, e = within & 15;
    const int half = lane >> 4;
    const int m = mt * 16 + (lane & 15);
    const int k = kt * 32 + a_kk_of(e, half);
    Wg_bf[tid] = (__bf16)Wg_w[(size_t)m * KX + k];
  }
  if (tid < MG * VP) {
    const int o = tid >> 5, v = tid & 31;
    bias_total[tid] = (v < VV) ? (Wg_b[o] + Ug_b[o] + bb[o * VV + v]) : 0.f;
  }
}

__global__ void zero_state(float* __restrict__ h_state, float* __restrict__ c_state) {
  const int tid = blockIdx.x * 256 + threadIdx.x;   // NB*HSZ*VP = 131072
  h_state[tid] = 0.f;
  c_state[tid] = 0.f;
}

// ---------------- K1a: sup_x for all timesteps (swizzled bf16 B) ------------
// supx layout: [t][kt(16)][ct(64)][512]   k = s*CIN + c, col = n*VP + v
__global__ void compute_supx(const float* __restrict__ x, const float* __restrict__ A_wg,
                             __bf16* __restrict__ supx) {
  const int tid  = blockIdx.x * 256 + threadIdx.x;  // 64*512*1024 = 1<<25
  const int t    = tid >> 19;                       // KX*COLS = 2^19 per t
  const int rem  = tid & ((1 << 19) - 1);
  const int tile = rem >> 9, within = rem & 511;
  const int kt = tile >> 6, ct = tile & 63;
  const int lane = within >> 4, e = within & 15;
  const int half = lane >> 4;
  const int k   = kt * 32 + b_kk_of(e, half);
  const int col = ct * 16 + (lane & 15);
  const int s = k >> 6, c = k & 63;
  const int n = col >> 5, v = col & 31;
  float acc = 0.f;
  if (v < VV) {
    const float* __restrict__ xr = x + (((size_t)(n * CINC + c)) * TT + t) * VV;
    const float* __restrict__ ar = A_wg + (s * VV + v) * VV;
#pragma unroll
    for (int u = 0; u < VV; ++u) acc += ar[u] * xr[u];
  }
  supx[tid] = (__bf16)acc;
}

// ---------------- K1b: gates_x = Wg @ sup_x for all t (WMMA) ----------------
__global__ void gemm_gates_x(const __bf16* __restrict__ Wg_bf,
                             const __bf16* __restrict__ supx,
                             float* __restrict__ gates_x) {
  const int W   = blockIdx.x * 8 + (threadIdx.x >> 5);  // global wave id
  const int t   = W >> 11;                              // 2048 tiles per t
  const int rem = W & 2047;
  const int mt  = rem >> 6;                             // 32 row tiles
  const int ct  = rem & 63;                             // 64 col tiles
  const __bf16* __restrict__ Bt = supx + ((size_t)t << 19);
  v8f acc = {0.f, 0.f, 0.f, 0.f, 0.f, 0.f, 0.f, 0.f};
  for (int kt = 0; kt < KX / 32; ++kt) {
    v16bf a = load_frag(Wg_bf, mt * (KX / 32) + kt);
    v16bf b = load_frag(Bt, kt * 64 + ct);
    acc = __builtin_amdgcn_wmma_f32_16x16x32_bf16(false, a, false, b,
                                                  (short)0, acc, false, false);
  }
  const int lane = threadIdx.x & 31;
  const int col  = ct * 16 + (lane & 15);
  const int half = lane >> 4;
  float* __restrict__ Dt = gates_x + (size_t)t * MG * COLS;
#pragma unroll
  for (int r = 0; r < 8; ++r) {
    const int m = mt * 16 + r + 8 * half;
    Dt[(size_t)m * COLS + col] = acc[r];
  }
}

// ---------------- K2: sup_h from current hidden state (swizzled) ------------
// sup_h layout: [kt(32)][ct(64)][512]   k = s*HS + c, col = n*VP + v
__global__ void compute_suph(const float* __restrict__ h_state,
                             const float* __restrict__ A_ug,
                             __bf16* __restrict__ sup_h) {
  const int tid  = blockIdx.x * 256 + threadIdx.x;  // KH*COLS = 1<<20
  const int tile = tid >> 9, within = tid & 511;
  const int kt = tile >> 6, ct = tile & 63;
  const int lane = within >> 4, e = within & 15;
  const int half = lane >> 4;
  const int k   = kt * 32 + b_kk_of(e, half);
  const int col = ct * 16 + (lane & 15);
  const int s = k >> 7, c = k & 127;
  const int n = col >> 5, v = col & 31;
  float acc = 0.f;
  if (v < VV) {
    const float* __restrict__ hr = h_state + (size_t)(n * HSZ + c) * VP;
    const float* __restrict__ ar = A_ug + (s * VV + v) * VV;
#pragma unroll
    for (int u = 0; u < VV; ++u) acc += ar[u] * hr[u];
  }
  sup_h[tid] = (__bf16)acc;
}

// ---------------- K3: recurrent GEMM (WMMA, LDS B-panel) + LSTM pointwise ---
// One block per 16 columns; wave wv owns hs rows [wv*16, wv*16+16) across all
// 4 gate groups (rows hs, hs+128, hs+256, hs+384) -> LSTM combine is in-wave.
__global__ void step_gemm_lstm(const __bf16* __restrict__ Ug_bf,
                               const __bf16* __restrict__ sup_h,
                               const float* __restrict__ gates_x_t,
                               const float* __restrict__ bias_total,
                               float* __restrict__ c_state,
                               float* __restrict__ h_pre) {
  __shared__ alignas(32) __bf16 Bs[KH * 16];        // 32 KB: 32 swizzled tiles
  const int ct = blockIdx.x;                        // column tile 0..63
  {
    // straight uint4 copy: 32 tiles x 64 uint4 each
    const uint4* __restrict__ src = (const uint4*)sup_h;
    uint4* dst = (uint4*)Bs;
    for (int i = threadIdx.x; i < 32 * 64; i += 256)
      dst[i] = src[(((i >> 6) * 64 + ct) << 6) + (i & 63)];
  }
  __syncthreads();

  const int wv   = threadIdx.x >> 5;
  const int lane = threadIdx.x & 31;
  const int hs0  = wv * 16;
  v8f accI = {0.f,0.f,0.f,0.f,0.f,0.f,0.f,0.f};
  v8f accF = accI, accG = accI, accO = accI;

  for (int kt = 0; kt < KH / 32; ++kt) {
    v16bf b  = load_frag_lds(Bs, kt);
    v16bf aI = load_frag(Ug_bf, (0 * 8 + wv) * (KH / 32) + kt);
    v16bf aF = load_frag(Ug_bf, (1 * 8 + wv) * (KH / 32) + kt);
    v16bf aG = load_frag(Ug_bf, (2 * 8 + wv) * (KH / 32) + kt);
    v16bf aO = load_frag(Ug_bf, (3 * 8 + wv) * (KH / 32) + kt);
    accI = __builtin_amdgcn_wmma_f32_16x16x32_bf16(false, aI, false, b, (short)0, accI, false, false);
    accF = __builtin_amdgcn_wmma_f32_16x16x32_bf16(false, aF, false, b, (short)0, accF, false, false);
    accG = __builtin_amdgcn_wmma_f32_16x16x32_bf16(false, aG, false, b, (short)0, accG, false, false);
    accO = __builtin_amdgcn_wmma_f32_16x16x32_bf16(false, aO, false, b, (short)0, accO, false, false);
  }

  const int col  = ct * 16 + (lane & 15);
  const int half = lane >> 4;
  const int nv   = col & (VP - 1);
  const int nn   = col >> 5;
#pragma unroll
  for (int r = 0; r < 8; ++r) {
    const int hs = hs0 + r + 8 * half;
    const float gi = accI[r] + gates_x_t[(size_t)(hs + 0 * HSZ) * COLS + col] + bias_total[(hs + 0 * HSZ) * VP + nv];
    const float gf = accF[r] + gates_x_t[(size_t)(hs + 1 * HSZ) * COLS + col] + bias_total[(hs + 1 * HSZ) * VP + nv];
    const float gg = accG[r] + gates_x_t[(size_t)(hs + 2 * HSZ) * COLS + col] + bias_total[(hs + 2 * HSZ) * VP + nv];
    const float go = accO[r] + gates_x_t[(size_t)(hs + 3 * HSZ) * COLS + col] + bias_total[(hs + 3 * HSZ) * VP + nv];
    const float iv = sigmoidf_(gi);
    const float fv = sigmoidf_(gf);
    const float gv = tanhf(gg);
    const float ov = sigmoidf_(go);
    const int sidx = (nn * HSZ + hs) * VP + nv;
    const float cn = fv * c_state[sidx] + iv * gv;
    c_state[sidx] = cn;
    h_pre[sidx]   = ov * tanhf(cn);
  }
}

// ---------------- K4: attention + residual, emit h and hidden_seq[t] --------
__global__ void attention_kernel(const float* __restrict__ h_pre,
                                 const float* __restrict__ Ww,  const float* __restrict__ Wb,
                                 const float* __restrict__ Wq,  const float* __restrict__ Wqb,
                                 const float* __restrict__ Wh,  const float* __restrict__ Whb,
                                 const float* __restrict__ Us,  const float* __restrict__ Usb,
                                 float* __restrict__ h_state, float* __restrict__ out_seq,
                                 int t) {
  __shared__ float sh_h[HSZ * VP];     // 16 KB
  __shared__ float sh_hsum[HSZ];
  __shared__ float sh_q[CC1];
  __shared__ float sh_qq[CC2];
  __shared__ float sh_u[VV * CC2];
  __shared__ float sh_a[VV];
  const int n   = blockIdx.x;
  const int tid = threadIdx.x;

  for (int i = tid; i < HSZ * VP; i += 256) sh_h[i] = h_pre[(size_t)n * HSZ * VP + i];
  __syncthreads();

  if (tid < HSZ) {
    float s = 0.f;
#pragma unroll
    for (int v = 0; v < VV; ++v) s += sh_h[tid * VP + v];
    sh_hsum[tid] = s;
  }
  __syncthreads();

  if (tid < CC1) {
    float s = Wb[tid];
    for (int hh = 0; hh < HSZ; ++hh) s += Ww[tid * HSZ + hh] * sh_hsum[hh];
    sh_q[tid] = s > 0.f ? s : 0.f;
  }
  __syncthreads();

  if (tid < CC2) {
    float s = Wqb[tid];
#pragma unroll
    for (int c = 0; c < CC1; ++c) s += Wq[tid * CC1 + c] * sh_q[c];
    sh_qq[tid] = s;
  }
  __syncthreads();

  for (int i = tid; i < VV * CC2; i += 256) {
    const int v = i / CC2, c2 = i % CC2;
    float s = Whb[c2] + sh_qq[c2];
    for (int hh = 0; hh < HSZ; ++hh) s += Wh[c2 * HSZ + hh] * sh_h[hh * VP + v];
    sh_u[i] = tanhf(s);
  }
  __syncthreads();

  if (tid < VV) {
    float s = Usb[0];
#pragma unroll
    for (int c2 = 0; c2 < CC2; ++c2) s += Us[c2] * sh_u[tid * CC2 + c2];
    sh_a[tid] = sigmoidf_(s);
  }
  __syncthreads();

  for (int i = tid; i < HSZ * VV; i += 256) {
    const int hh = i / VV, v = i % VV;
    const float hn = (1.f + sh_a[v]) * sh_h[hh * VP + v];
    h_state[(size_t)(n * HSZ + hh) * VP + v] = hn;
    out_seq[(((size_t)(n * HSZ + hh)) * TT + t) * VV + v] = hn;
  }
}

// ---------------- K5: hT / cT tail ------------------------------------------
__global__ void write_final(const float* __restrict__ h_state,
                            const float* __restrict__ c_state,
                            float* __restrict__ out) {
  const int tid = blockIdx.x * 256 + threadIdx.x;
  if (tid >= NB * HSZ * VV) return;
  const int n  = tid / (HSZ * VV);
  const int r  = tid % (HSZ * VV);
  const int hh = r / VV, v = r % VV;
  const size_t base1 = (size_t)NB * HSZ * TT * VV;
  out[base1 + tid]                         = h_state[(size_t)(n * HSZ + hh) * VP + v];
  out[base1 + (size_t)NB * HSZ * VV + tid] = c_state[(size_t)(n * HSZ + hh) * VP + v];
}

// ---------------- host launcher ---------------------------------------------
extern "C" void kernel_launch(void* const* d_in, const int* in_sizes, int n_in,
                              void* d_out, int out_size, void* d_ws, size_t ws_size,
                              hipStream_t stream) {
  const float* x    = (const float*)d_in[0];
  const float* A_wg = (const float*)d_in[1];
  const float* Wg_w = (const float*)d_in[2];
  const float* Wg_b = (const float*)d_in[3];
  const float* A_ug = (const float*)d_in[4];
  const float* Ug_w = (const float*)d_in[5];
  const float* Ug_b = (const float*)d_in[6];
  const float* bb   = (const float*)d_in[7];
  const float* Ww   = (const float*)d_in[8];
  const float* Wb   = (const float*)d_in[9];
  const float* Wq   = (const float*)d_in[10];
  const float* Wqb  = (const float*)d_in[11];
  const float* Wh   = (const float*)d_in[12];
  const float* Whb  = (const float*)d_in[13];
  const float* Us   = (const float*)d_in[14];
  const float* Usb  = (const float*)d_in[15];
  float* out = (float*)d_out;

  char* base = (char*)d_ws;
  size_t off = 0;
  auto alloc = [&](size_t bytes) -> void* {
    void* p = base + off;
    off = (off + bytes + 255) & ~(size_t)255;
    return p;
  };
  __bf16* Wg_bf      = (__bf16*)alloc((size_t)MG * KX * sizeof(__bf16));
  __bf16* Ug_bf      = (__bf16*)alloc((size_t)MG * KH * sizeof(__bf16));
  float*  bias_total = (float*) alloc((size_t)MG * VP * sizeof(float));
  __bf16* supx       = (__bf16*)alloc((size_t)TT * KX * COLS * sizeof(__bf16));   // 67 MB
  float*  gates_x    = (float*) alloc((size_t)TT * MG * COLS * sizeof(float));    // 134 MB
  __bf16* sup_h      = (__bf16*)alloc((size_t)KH * COLS * sizeof(__bf16));        // 2 MB
  float*  h_state    = (float*) alloc((size_t)NB * HSZ * VP * sizeof(float));
  float*  c_state    = (float*) alloc((size_t)NB * HSZ * VP * sizeof(float));
  float*  h_pre      = (float*) alloc((size_t)NB * HSZ * VP * sizeof(float));
  (void)ws_size; (void)in_sizes; (void)n_in; (void)out_size;

  // Phase 0: prep + state init
  prep_kernel<<<(MG * KH) / 256, 256, 0, stream>>>(Wg_w, Ug_w, Wg_b, Ug_b, bb,
                                                   Wg_bf, Ug_bf, bias_total);
  zero_state<<<(NB * HSZ * VP) / 256, 256, 0, stream>>>(h_state, c_state);

  // Phase 1: hoisted input GCN for all timesteps (parallel WMMA GEMM)
  compute_supx<<<(TT * KX * COLS) / 256, 256, 0, stream>>>(x, A_wg, supx);
  gemm_gates_x<<<(TT * 32 * 64) / 8, 256, 0, stream>>>(Wg_bf, supx, gates_x);

  // Phase 2: sequential recurrence
  for (int t = 0; t < TT; ++t) {
    compute_suph<<<(KH * COLS) / 256, 256, 0, stream>>>(h_state, A_ug, sup_h);
    step_gemm_lstm<<<COLS / 16, 256, 0, stream>>>(Ug_bf, sup_h,
                                                  gates_x + (size_t)t * MG * COLS,
                                                  bias_total, c_state, h_pre);
    attention_kernel<<<NB, 256, 0, stream>>>(h_pre, Ww, Wb, Wq, Wqb, Wh, Whb,
                                             Us, Usb, h_state, out, t);
  }

  // Phase 3: hT / cT
  write_final<<<(NB * HSZ * VV + 255) / 256, 256, 0, stream>>>(h_state, c_state, out);
}
